// GPT2_54176717472261
// MI455X (gfx1250) — matrix operand order repaired and checked by reading, not compiled, not run on hardware
//
#include <hip/hip_runtime.h>
#include <hip/hip_bf16.h>
#include <cstdint>

// ---------------------------------------------------------------------------
// GPT-2 forward for MI455X (gfx1250): bf16 WMMA + async global->LDS pipeline.
// ---------------------------------------------------------------------------

#define T_SEQ   2048
#define DMODEL  768
#define NLAYER  4
#define NHEAD   12
#define HSZ     64
#define DFF     3072
#define VOCAB   50257

typedef float  v8f   __attribute__((ext_vector_type(8)));
typedef __bf16 v16bf __attribute__((ext_vector_type(16)));

union ABu { v16bf v; uint4 q[2]; };

// fp32 -> bf16 round-to-nearest-even
__device__ __forceinline__ unsigned short f2bf(float f) {
  unsigned int u = __builtin_bit_cast(unsigned int, f);
  unsigned int r = u + 0x7FFFu + ((u >> 16) & 1u);
  return (unsigned short)(r >> 16);
}

// CDNA5 async copy: 16B global -> LDS, tracked by ASYNCcnt (no VGPR staging).
__device__ __forceinline__ void async_ld_b128(unsigned lds_off, const void* gptr) {
  asm volatile("global_load_async_to_lds_b128 %0, %1, off"
               :: "v"(lds_off), "v"((unsigned long long)(uintptr_t)gptr)
               : "memory");
}
__device__ __forceinline__ void wait_async0() {
  asm volatile("s_wait_asynccnt 0x0" ::: "memory");
}
__device__ __forceinline__ unsigned lds_off32(const void* p) {
  return (unsigned)(uintptr_t)p;   // LDS aperture: addr[31:0] is the LDS offset
}

// A operand (16x32 bf16) from row-major LDS tile, per CDNA5 VGPR layout:
// lane<16: M=lane, K={0..7,16..23}; lane>=16: M=lane-16, K={8..15,24..31}
__device__ __forceinline__ v16bf lds_load_a(const unsigned short* base, int stride, int lane) {
  int row  = lane & 15;
  int kofs = (lane & 16) ? 8 : 0;
  const unsigned short* p = base + row * stride + kofs;
  ABu u;
  u.q[0] = *(const uint4*)p;
  u.q[1] = *(const uint4*)(p + 16);
  return u.v;
}

// B operand (32x16 bf16) from *transposed* (N-major) LDS tile:
// lane<16: N=lane, K={0..7,16..23}; lane>=16: N=lane-16, K={8..15,24..31}
__device__ __forceinline__ v16bf lds_load_b(const unsigned short* base, int stride, int lane) {
  int col  = lane & 15;
  int kofs = (lane & 16) ? 8 : 0;
  const unsigned short* p = base + col * stride + kofs;
  ABu u;
  u.q[0] = *(const uint4*)p;
  u.q[1] = *(const uint4*)(p + 16);
  return u.v;
}

__device__ __forceinline__ v8f wmma_bf16(v16bf a, v16bf b, v8f c) {
  return __builtin_amdgcn_wmma_f32_16x16x32_bf16(false, a, false, b, (short)0, c, false, false);
}

// ---------------------------------------------------------------------------
// fp32 -> bf16 weight conversion (grid-stride)
// ---------------------------------------------------------------------------
__global__ void cvt_bf16_kernel(const float* __restrict__ in, unsigned short* __restrict__ out,
                                size_t n) {
  size_t i = (size_t)blockIdx.x * blockDim.x + threadIdx.x;
  size_t stride = (size_t)gridDim.x * blockDim.x;
  for (; i < n; i += stride) out[i] = f2bf(in[i]);
}

// ---------------------------------------------------------------------------
// embedding: x[t,:] = tok_emb[idx[t],:] + pos_emb[t,:]
// ---------------------------------------------------------------------------
__global__ void embed_kernel(const int* __restrict__ idx, const float* __restrict__ tok,
                             const float* __restrict__ pos, float* __restrict__ x) {
  int row = blockIdx.x;
  int tk  = idx[row];
  for (int d = threadIdx.x; d < DMODEL; d += blockDim.x)
    x[(size_t)row * DMODEL + d] = tok[(size_t)tk * DMODEL + d] + pos[(size_t)row * DMODEL + d];
}

// ---------------------------------------------------------------------------
// LayerNorm (row of DMODEL), output bf16 (feeds WMMA GEMM)
// ---------------------------------------------------------------------------
__global__ __launch_bounds__(256) void layernorm_kernel(const float* __restrict__ x,
                                                        const float* __restrict__ g,
                                                        const float* __restrict__ b,
                                                        unsigned short* __restrict__ y) {
  int row = blockIdx.x;
  const float* xr = x + (size_t)row * DMODEL;
  float s = 0.f, ss = 0.f;
  for (int d = threadIdx.x; d < DMODEL; d += 256) { float v = xr[d]; s += v; ss += v * v; }
  for (int o = 16; o > 0; o >>= 1) { s += __shfl_down(s, o, 32); ss += __shfl_down(ss, o, 32); }
  __shared__ float rs1[8], rs2[8];
  int wave = threadIdx.x >> 5, lane = threadIdx.x & 31;
  if (lane == 0) { rs1[wave] = s; rs2[wave] = ss; }
  __syncthreads();
  if (threadIdx.x == 0) {
    float a = 0.f, c = 0.f;
    for (int i = 0; i < 8; ++i) { a += rs1[i]; c += rs2[i]; }
    rs1[0] = a; rs2[0] = c;
  }
  __syncthreads();
  float mean = rs1[0] * (1.0f / DMODEL);
  float var  = rs2[0] * (1.0f / DMODEL) - mean * mean;
  float rstd = rsqrtf(var + 1e-5f);
  for (int d = threadIdx.x; d < DMODEL; d += 256)
    y[(size_t)row * DMODEL + d] = f2bf((xr[d] - mean) * rstd * g[d] + b[d]);
}

// ---------------------------------------------------------------------------
// Tiled bf16 WMMA GEMM: C[M,N] = A[M,K] * B[K,N] (+bias)(+residual)(relu)
// BM=128, BN=128, BK=32. 256 threads = 8 waves, wave tile 32x64 (8 accums).
// Double-buffered LDS; A tiles stream via GLOBAL_LOAD_ASYNC_TO_LDS_B128,
// B tiles are VGPR-staged (transpose) with global reads hoisted one tile early.
// M % 128 == 0 and K % 32 == 0 guaranteed by callers; N is bounds-checked.
// ---------------------------------------------------------------------------
#define BM 128
#define BN 128
#define BK 32

template <bool BIAS, bool RES, bool RELU, bool WF32, bool WBF16>
__global__ __launch_bounds__(256) void gemm_bf16_kernel(
    const unsigned short* __restrict__ A,   // M x K bf16
    const unsigned short* __restrict__ B,   // K x N bf16
    const float* bias,                      // N
    const float* res,                       // M x N (may alias Cf)
    float* Cf,                              // M x N
    unsigned short* Cb,                     // M x N
    int M, int N, int K) {
  __shared__ __align__(16) unsigned short sA[2][BM * BK];    // row-major, 2x8KB
  __shared__ __align__(16) unsigned short sBT[2][BN * BK];   // N-major,   2x8KB
  int t = threadIdx.x, wave = t >> 5, lane = t & 31;
  int wm = wave & 3, wn = wave >> 2;
  int m0 = blockIdx.y * BM;
  int n0 = blockIdx.x * BN;
  int nt = K / BK;
  v8f acc[2][4] = {};

  // --- async A-tile issue: 128x32 bf16 = 512 x 16B chunks, 2 per thread ---
  auto issueA = [&](int tile, int buf) {
    unsigned base = lds_off32(&sA[buf][0]);
    int k0 = tile * BK;
    #pragma unroll
    for (int i = 0; i < 2; ++i) {
      int c = t + 256 * i;
      int row = c >> 2, kc = (c & 3) * 8;
      async_ld_b128(base + c * 16, A + (size_t)(m0 + row) * K + k0 + kc);
    }
  };
  // --- B-tile global read into regs (coalesced), 16 elems/thread ---
  unsigned short breg[16];
  auto loadB = [&](int tile) {
    int k0 = tile * BK;
    #pragma unroll
    for (int i = 0; i < 16; ++i) {
      int e = t + 256 * i;
      int kl = e >> 7, nl = e & 127;
      int n = n0 + nl;
      breg[i] = (n < N) ? B[(size_t)(k0 + kl) * N + n] : (unsigned short)0;
    }
  };
  auto storeBT = [&](int buf) {
    #pragma unroll
    for (int i = 0; i < 16; ++i) {
      int e = t + 256 * i;
      int kl = e >> 7, nl = e & 127;
      sBT[buf][nl * BK + kl] = breg[i];
    }
  };

  // prologue: stage tile 0
  issueA(0, 0);
  loadB(0);
  storeBT(0);

  for (int i = 0; i < nt; ++i) {
    int cur = i & 1, nxt = cur ^ 1;
    if (i + 1 < nt) loadB(i + 1);          // global reads overlap compute
    wait_async0();                          // A(cur) landed in LDS
    __syncthreads();                        // all waves past reads of buffers[nxt]
    if (i + 1 < nt) { issueA(i + 1, nxt); storeBT(nxt); }

    const unsigned short* abase = &sA[cur][(wm * 32) * BK];
    v16bf av0 = lds_load_a(abase, BK, lane);
    v16bf av1 = lds_load_a(abase + 16 * BK, BK, lane);
    #pragma unroll
    for (int j = 0; j < 4; ++j) {
      v16bf bv = lds_load_b(&sBT[cur][(wn * 64 + j * 16) * BK], BK, lane);
      acc[0][j] = wmma_bf16(av0, bv, acc[0][j]);
      acc[1][j] = wmma_bf16(av1, bv, acc[1][j]);
    }
  }

  // epilogue; C layout: VGPR r, lane -> (M = r + 8*(lane>>4), N = lane&15)
  int mbase = m0 + wm * 32 + 8 * (lane >> 4);
  int nbase = n0 + wn * 64 + (lane & 15);
  #pragma unroll
  for (int a = 0; a < 2; ++a) {
    #pragma unroll
    for (int j = 0; j < 4; ++j) {
      int n = nbase + j * 16;
      if (n >= N) continue;
      float bv = BIAS ? bias[n] : 0.f;
      #pragma unroll
      for (int r = 0; r < 8; ++r) {
        size_t off = (size_t)(mbase + a * 16 + r) * N + n;
        float v = acc[a][j][r] + bv;
        if (RES)  v += res[off];
        if (RELU) v = fmaxf(v, 0.f);
        if (WF32)  Cf[off] = v;
        if (WBF16) Cb[off] = f2bf(v);
      }
    }
  }
}

// ---------------------------------------------------------------------------
// Flash attention: 1 workgroup = (head, 64-query block), 4 waves x 16 queries.
// Q/K tiles stream via async->LDS; V tiles VGPR-staged (transpose), K/V
// double-buffered. q/k/v/o are [T, DMODEL] bf16, head slice = cols head*64..+63.
// ---------------------------------------------------------------------------
__global__ __launch_bounds__(128) void attn_kernel(const unsigned short* __restrict__ q,
                                                   const unsigned short* __restrict__ k,
                                                   const unsigned short* __restrict__ v,
                                                   unsigned short* __restrict__ o) {
  int head  = blockIdx.x;
  int qbase = blockIdx.y * 64;
  int hofs  = head * HSZ;
  int t = threadIdx.x, wave = t >> 5, lane = t & 31;

  __shared__ __align__(16) unsigned short sQ[64 * 64];         // [query][dim] 8KB
  __shared__ __align__(16) unsigned short sK[2][32 * 64];      // [key][dim]   2x4KB
  __shared__ __align__(16) unsigned short sVT[2][64 * 32];     // [dim][key]   2x4KB
  __shared__ __align__(16) unsigned short sP[4][16 * 32];      // per-wave P   4KB

  // async Q tile: 64x64 = 512 x 16B chunks, 4 per thread
  {
    unsigned base = lds_off32(&sQ[0]);
    #pragma unroll
    for (int i = 0; i < 4; ++i) {
      int c = t + 128 * i;
      int row = c >> 3, d8 = (c & 7) * 8;
      async_ld_b128(base + c * 16, q + (size_t)(qbase + row) * DMODEL + hofs + d8);
    }
  }
  auto issueK = [&](int blk, int buf) {   // 32x64 = 256 chunks, 2 per thread
    unsigned base = lds_off32(&sK[buf][0]);
    #pragma unroll
    for (int i = 0; i < 2; ++i) {
      int c = t + 128 * i;
      int row = c >> 3, d8 = (c & 7) * 8;
      async_ld_b128(base + c * 16, k + (size_t)(blk * 32 + row) * DMODEL + hofs + d8);
    }
  };
  unsigned short vreg[16];
  auto loadV = [&](int blk) {
    #pragma unroll
    for (int i = 0; i < 16; ++i) {
      int e = t + 128 * i;
      int kl = e >> 6, d = e & 63;
      vreg[i] = v[(size_t)(blk * 32 + kl) * DMODEL + hofs + d];
    }
  };
  auto storeVT = [&](int buf) {
    #pragma unroll
    for (int i = 0; i < 16; ++i) {
      int e = t + 128 * i;
      int kl = e >> 6, d = e & 63;
      sVT[buf][d * 32 + kl] = vreg[i];
    }
  };

  int nblocks = 2 * (blockIdx.y + 1);                // keys 0 .. qbase+63
  issueK(0, 0);
  loadV(0);
  storeVT(0);

  v8f acc[4] = {};
  float mrow[8], lrow[8];
  #pragma unroll
  for (int r = 0; r < 8; ++r) { mrow[r] = -1e30f; lrow[r] = 0.f; }

  int mglo  = qbase + wave * 16 + 8 * (lane >> 4);   // + r = global query row
  int qw_hi = qbase + wave * 16 + 15;

  for (int kb = 0; kb < nblocks; ++kb) {
    int kbase = kb * 32;
    int cur = kb & 1, nxt = cur ^ 1;
    if (kb + 1 < nblocks) loadV(kb + 1);
    wait_async0();                                   // Q (kb==0) and K(cur) landed
    __syncthreads();
    if (kb + 1 < nblocks) { issueK(kb + 1, nxt); storeVT(nxt); }

    if (kbase <= qw_hi) {                            // wave-uniform causal skip
      // S = Q(16x64) * K^T(64x32): two 16x16 tiles, 2 WMMA each
      v8f s0 = {}, s1 = {};
      #pragma unroll
      for (int kk = 0; kk < 2; ++kk) {
        v16bf aq = lds_load_a(sQ + wave * 16 * 64 + kk * 32, 64, lane);
        v16bf b0 = lds_load_b(&sK[cur][0 * 64 + kk * 32], 64, lane);
        v16bf b1 = lds_load_b(&sK[cur][16 * 64 + kk * 32], 64, lane);
        s0 = wmma_bf16(aq, b0, s0);
        s1 = wmma_bf16(aq, b1, s1);
      }
      // online softmax (scale 1/sqrt(64)=0.125, causal mask)
      int n = lane & 15;
      #pragma unroll
      for (int r = 0; r < 8; ++r) {
        int mg = mglo + r;
        float e0 = s0[r] * 0.125f, e1 = s1[r] * 0.125f;
        if (kbase + n > mg)      e0 = -1e30f;
        if (kbase + 16 + n > mg) e1 = -1e30f;
        float mx = fmaxf(e0, e1);
        for (int o2 = 8; o2 > 0; o2 >>= 1) mx = fmaxf(mx, __shfl_xor(mx, o2, 32));
        float mn = fmaxf(mrow[r], mx);
        float alpha = __expf(mrow[r] - mn);
        float p0 = __expf(e0 - mn), p1 = __expf(e1 - mn);
        float rsum = p0 + p1;
        for (int o2 = 8; o2 > 0; o2 >>= 1) rsum += __shfl_xor(rsum, o2, 32);
        lrow[r] = lrow[r] * alpha + rsum;
        mrow[r] = mn;
        #pragma unroll
        for (int j = 0; j < 4; ++j) acc[j][r] *= alpha;
        int pm = r + 8 * (lane >> 4);
        sP[wave][pm * 32 + n]      = f2bf(p0);
        sP[wave][pm * 32 + 16 + n] = f2bf(p1);
      }
      // make per-wave P writes visible to A-operand loads (CDNA5 split counter)
      asm volatile("s_wait_dscnt 0x0" ::: "memory");
      // acc += P(16x32) * V(32x64)
      v16bf ap = lds_load_a(sP[wave], 32, lane);
      #pragma unroll
      for (int j = 0; j < 4; ++j) {
        v16bf bv = lds_load_b(&sVT[cur][(j * 16) * 32], 32, lane);
        acc[j] = wmma_bf16(ap, bv, acc[j]);
      }
    }
  }

  // write normalized output (bf16, feeds Wo GEMM)
  #pragma unroll
  for (int j = 0; j < 4; ++j) {
    int n = hofs + j * 16 + (lane & 15);
    #pragma unroll
    for (int r = 0; r < 8; ++r) {
      int m = mglo + r;
      o[(size_t)m * DMODEL + n] = f2bf(acc[j][r] / lrow[r]);
    }
  }
}

// ---------------------------------------------------------------------------
// host orchestration
// ---------------------------------------------------------------------------
extern "C" void kernel_launch(void* const* d_in, const int* in_sizes, int n_in,
                              void* d_out, int out_size, void* d_ws, size_t ws_size,
                              hipStream_t stream) {
  const int*   idx    = (const int*)  d_in[0];
  const float* tok    = (const float*)d_in[1];
  const float* pos    = (const float*)d_in[2];
  const float* ln1_g  = (const float*)d_in[3];
  const float* ln1_b  = (const float*)d_in[4];
  const float* Wq     = (const float*)d_in[5];
  const float* Wk     = (const float*)d_in[6];
  const float* Wv     = (const float*)d_in[7];
  const float* Wo     = (const float*)d_in[8];
  const float* bo     = (const float*)d_in[9];
  const float* ln2_g  = (const float*)d_in[10];
  const float* ln2_b  = (const float*)d_in[11];
  const float* W1     = (const float*)d_in[12];
  const float* b1     = (const float*)d_in[13];
  const float* W2     = (const float*)d_in[14];
  const float* b2     = (const float*)d_in[15];
  const float* lnf_g  = (const float*)d_in[16];
  const float* lnf_b  = (const float*)d_in[17];
  const float* lm_W   = (const float*)d_in[18];
  const float* lm_b   = (const float*)d_in[19];
  float* out = (float*)d_out;

  // workspace carve-out (~169 MB total)
  char* p = (char*)d_ws;
  auto alloc = [&](size_t bytes) -> void* {
    void* r = (void*)p;
    p += (bytes + 255) & ~(size_t)255;
    return r;
  };
  float*          x    = (float*)         alloc((size_t)T_SEQ * DMODEL * 4);
  unsigned short* hb   = (unsigned short*)alloc((size_t)T_SEQ * DMODEL * 2);
  unsigned short* qb   = (unsigned short*)alloc((size_t)T_SEQ * DMODEL * 2);
  unsigned short* kb   = (unsigned short*)alloc((size_t)T_SEQ * DMODEL * 2);
  unsigned short* vb   = (unsigned short*)alloc((size_t)T_SEQ * DMODEL * 2);
  unsigned short* ab   = (unsigned short*)alloc((size_t)T_SEQ * DMODEL * 2);
  unsigned short* hid  = (unsigned short*)alloc((size_t)T_SEQ * DFF * 2);
  unsigned short* wqb  = (unsigned short*)alloc((size_t)NLAYER * DMODEL * DMODEL * 2);
  unsigned short* wkb  = (unsigned short*)alloc((size_t)NLAYER * DMODEL * DMODEL * 2);
  unsigned short* wvb  = (unsigned short*)alloc((size_t)NLAYER * DMODEL * DMODEL * 2);
  unsigned short* wob  = (unsigned short*)alloc((size_t)NLAYER * DMODEL * DMODEL * 2);
  unsigned short* w1b  = (unsigned short*)alloc((size_t)NLAYER * DMODEL * DFF * 2);
  unsigned short* w2b  = (unsigned short*)alloc((size_t)NLAYER * DFF * DMODEL * 2);
  unsigned short* lmwb = (unsigned short*)alloc((size_t)DMODEL * VOCAB * 2);

  const size_t dd = (size_t)DMODEL * DMODEL;
  const size_t df = (size_t)DMODEL * DFF;

  // one-shot weight conversions (deterministic, graph-capture safe)
  cvt_bf16_kernel<<<2048, 256, 0, stream>>>(Wq, wqb, NLAYER * dd);
  cvt_bf16_kernel<<<2048, 256, 0, stream>>>(Wk, wkb, NLAYER * dd);
  cvt_bf16_kernel<<<2048, 256, 0, stream>>>(Wv, wvb, NLAYER * dd);
  cvt_bf16_kernel<<<2048, 256, 0, stream>>>(Wo, wob, NLAYER * dd);
  cvt_bf16_kernel<<<2048, 256, 0, stream>>>(W1, w1b, NLAYER * df);
  cvt_bf16_kernel<<<2048, 256, 0, stream>>>(W2, w2b, NLAYER * df);
  cvt_bf16_kernel<<<2048, 256, 0, stream>>>(lm_W, lmwb, (size_t)DMODEL * VOCAB);

  embed_kernel<<<T_SEQ, 256, 0, stream>>>(idx, tok, pos, x);

  dim3 gD(DMODEL / BN, T_SEQ / BM);   // (6, 16)
  dim3 gF(DFF / BN,    T_SEQ / BM);   // (24, 16)
  dim3 gV((VOCAB + BN - 1) / BN, T_SEQ / BM);
  dim3 gA(NHEAD, T_SEQ / 64);

  for (int L = 0; L < NLAYER; ++L) {
    // ln1 -> h (bf16)
    layernorm_kernel<<<T_SEQ, 256, 0, stream>>>(x, ln1_g + L * DMODEL, ln1_b + L * DMODEL, hb);
    // q/k/v = h @ W{q,k,v}  (bf16 out)
    gemm_bf16_kernel<false, false, false, false, true><<<gD, 256, 0, stream>>>(
        hb, wqb + L * dd, nullptr, nullptr, nullptr, qb, T_SEQ, DMODEL, DMODEL);
    gemm_bf16_kernel<false, false, false, false, true><<<gD, 256, 0, stream>>>(
        hb, wkb + L * dd, nullptr, nullptr, nullptr, kb, T_SEQ, DMODEL, DMODEL);
    gemm_bf16_kernel<false, false, false, false, true><<<gD, 256, 0, stream>>>(
        hb, wvb + L * dd, nullptr, nullptr, nullptr, vb, T_SEQ, DMODEL, DMODEL);
    // causal flash attention
    attn_kernel<<<gA, 128, 0, stream>>>(qb, kb, vb, ab);
    // x = x + attn @ Wo + bo  (f32 out, residual)
    gemm_bf16_kernel<true, true, false, true, false><<<gD, 256, 0, stream>>>(
        ab, wob + L * dd, bo + L * DMODEL, x, x, nullptr, T_SEQ, DMODEL, DMODEL);
    // ln2 -> h (bf16)
    layernorm_kernel<<<T_SEQ, 256, 0, stream>>>(x, ln2_g + L * DMODEL, ln2_b + L * DMODEL, hb);
    // hid = relu(h @ W1 + b1)  (bf16 out)
    gemm_bf16_kernel<true, false, true, false, true><<<gF, 256, 0, stream>>>(
        hb, w1b + L * df, b1 + L * DFF, nullptr, nullptr, hid, T_SEQ, DFF, DMODEL);
    // x = x + hid @ W2 + b2  (f32 out, residual)
    gemm_bf16_kernel<true, true, false, true, false><<<gD, 256, 0, stream>>>(
        hid, w2b + L * df, b2 + L * DMODEL, x, x, nullptr, T_SEQ, DMODEL, DFF);
  }

  // final LN + LM head
  layernorm_kernel<<<T_SEQ, 256, 0, stream>>>(x, lnf_g, lnf_b, hb);
  gemm_bf16_kernel<true, false, false, true, false><<<gV, 256, 0, stream>>>(
      hb, lmwb, lm_b, nullptr, out, nullptr, T_SEQ, VOCAB, DMODEL);
}
